// EnhancedSTGraphNet_31361851195620
// MI455X (gfx1250) — matrix operand order repaired and checked by reading, not compiled, not run on hardware
//
#include <hip/hip_runtime.h>
#include <hip/hip_bf16.h>
#include <math.h>

// ---------------------------------------------------------------------------
// EnhancedSTGraphNet on MI455X (gfx1250, wave32, WMMA).
// Dense layers: bf16 v_wmma_f32_16x16x32_bf16, f32 accum, fused bias/LN/GELU.
//   - weights pre-converted once to zero-padded bf16 images in workspace
//   - W slabs staged to LDS via global_load_async_to_lds_b128 (ASYNCcnt)
//   - A fragments loaded as 4x b128 + packed bf16 cvt
// Sparse props / attention: L2-resident gather + f32 atomic scatter.
// Workspace demand: ~57.5M floats (~230 MB).
// ---------------------------------------------------------------------------

typedef __bf16 bf16_t;
typedef __attribute__((ext_vector_type(16))) __bf16 bf16x16;
typedef __attribute__((ext_vector_type(8)))  float  f32x8;

static constexpr int NN = 50000;   // nodes
static constexpr int NE = 800000;  // edges
static constexpr int DD = 192;     // hidden dim
static constexpr int NH = 8;       // heads
static constexpr int CC = 24;      // channels per head
static constexpr int NG = 64;      // graphs

// GEMM epilogue flags
#define GF_LN   1
#define GF_GELU 2
#define GF_ACC  4
// row-op flags (applied in this order: MUL, BIAS, LN, RES, GELU)
#define RF_BIAS 1
#define RF_LN   2
#define RF_RES  4
#define RF_GELU 8
#define RF_MUL  16

__device__ __forceinline__ float gelu_exact(float v) {
    return 0.5f * v * (1.0f + erff(v * 0.70710678118654752f));
}
__device__ __forceinline__ float lrelu02(float v) { return v > 0.f ? v : 0.2f * v; }

// order-preserving float<->uint encoding for atomic max
__device__ __forceinline__ unsigned fenc(float f) {
    unsigned u = __float_as_uint(f);
    return (u & 0x80000000u) ? ~u : (u | 0x80000000u);
}
__device__ __forceinline__ float fdec(unsigned e) {
    unsigned u = (e & 0x80000000u) ? (e & 0x7FFFFFFFu) : ~e;
    return __uint_as_float(u);
}
#define ENC_NEG_INF 0x007FFFFFu   // fenc(-inf)

// ---------------------------------------------------------------------------
// weight pre-conversion: f32 (K x DOUT) -> bf16 (Kpad x DOUT), zero-padded
// ---------------------------------------------------------------------------
__global__ void k_cvtw(const float* __restrict__ W, bf16_t* dst,
                       int K, int Kpad, int DOUT)
{
    long idx = (long)blockIdx.x * 256 + threadIdx.x;
    if (idx >= (long)Kpad * DOUT) return;
    int kk = (int)(idx / DOUT), n = (int)(idx % DOUT);
    dst[idx] = (kk < K) ? (bf16_t)W[(long)kk * DOUT + n] : (bf16_t)0.f;
}

// ---------------------------------------------------------------------------
// Fused GEMM: out[nrows x DOUT] = epilogue(A[nrows x K] @ W[Kpad x DOUT] + bias)
// DOUT = NT*16; W is pre-converted padded bf16.  Block = 256 threads = 8
// wave32; each wave owns 16 rows and the full DOUT strip (NT WMMA tiles).
// W slab (32 x DOUT bf16) staged to LDS with async global->LDS b128 copies.
// C/D layout (ISA 7.12.2): lane L holds rows {8*(L/16)+r}, col = t*16 + L%16.
// A 16x32 bf16 layout: lane L = row L%16, element j -> k = 16*(j/8)+8*(L/16)+(j%8).
// B 32x16 bf16 layout: lane L = col L%16, element j -> same k as A.
// ---------------------------------------------------------------------------
template <int NT>
__global__ __launch_bounds__(256)
void k_gemm(const float* __restrict__ A, const bf16_t* __restrict__ W,
            const float* __restrict__ bias, float* out,
            int nrows, int K, int Kpad, int flags)
{
    constexpr int DOUT = NT * 16;
    __shared__ alignas(32) bf16_t wlds[32 * 192];

    const int tid  = threadIdx.x;
    const int wave = tid >> 5;
    const int lane = tid & 31;
    const int half = lane >> 4;
    const int lm   = lane & 15;
    const int rowbase = blockIdx.x * 128 + wave * 16;
    const unsigned ldsbase = (unsigned)(size_t)(void*)&wlds[0];

    f32x8 acc[NT];
#pragma unroll
    for (int t = 0; t < NT; ++t) {
        float bv = 0.f;
        if (!(flags & GF_ACC) && bias) bv = bias[t * 16 + lm];
#pragma unroll
        for (int r = 0; r < 8; ++r) acc[t][r] = bv;
    }
    if (flags & GF_ACC) {
#pragma unroll
        for (int t = 0; t < NT; ++t)
#pragma unroll
            for (int r = 0; r < 8; ++r) {
                int row = rowbase + r + 8 * half;
                if (row < nrows) acc[t][r] = out[(long)row * DOUT + t * 16 + lm];
            }
    }

    const int  arow   = rowbase + lm;
    const bool avalid = arow < nrows;

    for (int k0 = 0; k0 < Kpad; k0 += 32) {
        // --- stage W slab (32 x DOUT bf16 = 64*DOUT bytes) via async copy ---
        {
            const char* wsrc = (const char*)(W + (size_t)k0 * DOUT);
            for (int b = tid * 16; b < 64 * DOUT; b += 4096) {
                unsigned loff = ldsbase + (unsigned)b;
                asm volatile("global_load_async_to_lds_b128 %0, %1, off"
                             :: "v"(loff), "v"(wsrc + b) : "memory");
            }
            asm volatile("s_wait_asynccnt 0x0" ::: "memory");
        }
        __syncthreads();

        if (avalid && (k0 + 32 < Kpad))
            __builtin_prefetch(&A[(long)arow * K + k0 + 32], 0, 3);

        // --- A fragment ---
        bf16x16 afrag;
        if (avalid && (k0 + 32 <= K)) {
            const float4* p0 = reinterpret_cast<const float4*>(&A[(long)arow * K + k0 + half * 8]);
            const float4* p1 = reinterpret_cast<const float4*>(&A[(long)arow * K + k0 + 16 + half * 8]);
            float4 a0 = p0[0], a1 = p0[1];
            float4 b0 = p1[0], b1 = p1[1];
            afrag[0]  = (bf16_t)a0.x; afrag[1]  = (bf16_t)a0.y;
            afrag[2]  = (bf16_t)a0.z; afrag[3]  = (bf16_t)a0.w;
            afrag[4]  = (bf16_t)a1.x; afrag[5]  = (bf16_t)a1.y;
            afrag[6]  = (bf16_t)a1.z; afrag[7]  = (bf16_t)a1.w;
            afrag[8]  = (bf16_t)b0.x; afrag[9]  = (bf16_t)b0.y;
            afrag[10] = (bf16_t)b0.z; afrag[11] = (bf16_t)b0.w;
            afrag[12] = (bf16_t)b1.x; afrag[13] = (bf16_t)b1.y;
            afrag[14] = (bf16_t)b1.z; afrag[15] = (bf16_t)b1.w;
        } else {
#pragma unroll
            for (int j = 0; j < 16; ++j) {
                int k = k0 + ((j >> 3) << 4) + (half << 3) + (j & 7);
                float v = (avalid && k < K) ? A[(long)arow * K + k] : 0.f;
                afrag[j] = (bf16_t)v;
            }
        }

#pragma unroll
        for (int t = 0; t < NT; ++t) {
            bf16x16 bfrag;
#pragma unroll
            for (int j = 0; j < 16; ++j)
                bfrag[j] = wlds[(((j >> 3) << 4) + (half << 3) + (j & 7)) * DOUT + t * 16 + lm];
            acc[t] = __builtin_amdgcn_wmma_f32_16x16x32_bf16(
                false, afrag, false, bfrag, (short)0, acc[t], false, false);
        }
        __syncthreads();
    }

    // epilogue: optional LayerNorm (row lives in 16 lanes of this half) + GELU
    float mean[8], rstd[8];
    if (flags & GF_LN) {
#pragma unroll
        for (int r = 0; r < 8; ++r) {
            float s = 0.f, q = 0.f;
#pragma unroll
            for (int t = 0; t < NT; ++t) { float v = acc[t][r]; s += v; q += v * v; }
#pragma unroll
            for (int m = 8; m >= 1; m >>= 1) {  // stays within the 16-lane half
                s += __shfl_xor(s, m, 32);
                q += __shfl_xor(q, m, 32);
            }
            float mu  = s / (float)DOUT;
            float var = q / (float)DOUT - mu * mu;
            mean[r] = mu;
            rstd[r] = rsqrtf(var + 1e-5f);
        }
    }
#pragma unroll
    for (int t = 0; t < NT; ++t)
#pragma unroll
        for (int r = 0; r < 8; ++r) {
            float v = acc[t][r];
            if (flags & GF_LN)   v = (v - mean[r]) * rstd[r];
            if (flags & GF_GELU) v = gelu_exact(v);
            int row = rowbase + r + 8 * half;
            if (row < nrows) out[(long)row * DOUT + t * 16 + lm] = v;
        }
}

// ---------------------------------------------------------------------------
// Generic per-row elementwise ops: one wave32 per row (width <= 192).
// x = in; [MUL] x*=g[row]; [BIAS] x+=b[d]; [LN] layernorm(x); [RES] x+=res; [GELU]
// ---------------------------------------------------------------------------
__global__ __launch_bounds__(256)
void k_rowops(const float* __restrict__ in, float* out,
              const float* __restrict__ bias, const float* __restrict__ rowscale,
              const float* __restrict__ res, int n, int width, int flags)
{
    int row  = blockIdx.x * 8 + (threadIdx.x >> 5);
    int lane = threadIdx.x & 31;
    if (row >= n) return;
    float g = (flags & RF_MUL) ? rowscale[row] : 1.f;
    float x[6];
#pragma unroll
    for (int i = 0; i < 6; ++i) {
        int d = lane + i * 32;
        float v = 0.f;
        if (d < width) {
            v = in[(long)row * width + d] * g;
            if (flags & RF_BIAS) v += bias[d];
        }
        x[i] = v;
    }
    if (flags & RF_LN) {
        float s = 0.f, q = 0.f;
#pragma unroll
        for (int i = 0; i < 6; ++i) { s += x[i]; q += x[i] * x[i]; }
#pragma unroll
        for (int m = 16; m >= 1; m >>= 1) { s += __shfl_xor(s, m, 32); q += __shfl_xor(q, m, 32); }
        float mu = s / (float)width;
        float rs = rsqrtf(q / (float)width - mu * mu + 1e-5f);
#pragma unroll
        for (int i = 0; i < 6; ++i) x[i] = (x[i] - mu) * rs;
    }
#pragma unroll
    for (int i = 0; i < 6; ++i) {
        int d = lane + i * 32;
        if (d >= width) continue;
        float v = x[i];
        if (flags & RF_RES)  v += res[(long)row * width + d];
        if (flags & RF_GELU) v = gelu_exact(v);
        out[(long)row * width + d] = v;
    }
}

// -------------------------- small utility kernels --------------------------
__global__ void k_fill_f32(float* p, float v, long n)
{ long i = (long)blockIdx.x * 256 + threadIdx.x; if (i < n) p[i] = v; }
__global__ void k_fill_u32(unsigned* p, unsigned v, long n)
{ long i = (long)blockIdx.x * 256 + threadIdx.x; if (i < n) p[i] = v; }

// GCN normalization
__global__ void k_deg(float* deg, const int* col, const float* w, int E)
{ int e = blockIdx.x * 256 + threadIdx.x; if (e < E) atomicAdd(&deg[col[e]], w[e]); }
__global__ void k_dinv(float* deg, int n)
{ int i = blockIdx.x * 256 + threadIdx.x; if (i < n) deg[i] = rsqrtf(deg[i] + 1.0f); }
__global__ void k_norm(float* norm, const int* row, const int* col,
                       const float* w, const float* dinv, int E)
{ int e = blockIdx.x * 256 + threadIdx.x;
  if (e < E) norm[e] = dinv[row[e]] * w[e] * dinv[col[e]]; }

// prop: out = self_loop_term; then out[col] += feat[row]*norm (atomic)
__global__ void k_prop_init(float* out, const float* feat, const float* dinv,
                            int n, int width)
{ long idx = (long)blockIdx.x * 256 + threadIdx.x;
  if (idx < (long)n * width) { int i = (int)(idx / width); float dv = dinv[i];
      out[idx] = feat[idx] * dv * dv; } }
__global__ void k_prop_edges(float* out, const float* __restrict__ feat,
                             const int* __restrict__ row, const int* __restrict__ col,
                             const float* __restrict__ norm, int width)
{ int e = blockIdx.x; int d = threadIdx.x;
  atomicAdd(&out[(long)col[e] * width + d], feat[(long)row[e] * width + d] * norm[e]); }

// per-(node,head) dots for GAT
__global__ void k_head_dots(const float* __restrict__ xp, const float* __restrict__ as,
                            const float* __restrict__ ad, float* asrc, float* adst, int n)
{ int idx = blockIdx.x * 256 + threadIdx.x; if (idx >= n * NH) return;
  int i = idx / NH, h = idx % NH; float s1 = 0.f, s2 = 0.f;
  for (int c = 0; c < CC; ++c) { float v = xp[(long)i * DD + h * CC + c];
      s1 += v * as[h * CC + c]; s2 += v * ad[h * CC + c]; }
  asrc[idx] = s1; adst[idx] = s2; }

__global__ void k_aedot(const float* eW, const float* ae, float* aedot)
{ int h = threadIdx.x; if (h >= NH) return; float s = 0.f;
  for (int c = 0; c < CC; ++c) s += eW[h * CC + c] * ae[h * CC + c];
  aedot[h] = s; }

// edge attention: logits + segment max
__global__ void k_edge_logits(const float* __restrict__ asrc, const float* __restrict__ adst,
                              const float* __restrict__ eattr, const float* __restrict__ aedot,
                              const int* __restrict__ row, const int* __restrict__ col,
                              const int* __restrict__ seg, unsigned* mws, float* logits,
                              int E, int Hn)
{ long idx = (long)blockIdx.x * 256 + threadIdx.x; if (idx >= (long)E * Hn) return;
  int e = (int)(idx / Hn), h = (int)(idx % Hn);
  float et = aedot ? eattr[e] * aedot[h] : 0.f;
  float l = lrelu02(asrc[(long)row[e] * Hn + h] + adst[(long)col[e] * Hn + h] + et);
  logits[idx] = l;
  atomicMax(&mws[(long)seg[e] * Hn + h], fenc(l)); }

__global__ void k_edge_exp(float* logits, const unsigned* __restrict__ mws,
                           const int* __restrict__ seg, float* sws, int E, int Hn)
{ long idx = (long)blockIdx.x * 256 + threadIdx.x; if (idx >= (long)E * Hn) return;
  int e = (int)(idx / Hn), h = (int)(idx % Hn);
  float v = expf(logits[idx] - fdec(mws[(long)seg[e] * Hn + h]));
  logits[idx] = v;
  atomicAdd(&sws[(long)seg[e] * Hn + h], v); }

__global__ void k_edge_agg(const float* __restrict__ feat, const float* __restrict__ logits,
                           const float* __restrict__ sws, const int* __restrict__ gidx,
                           const int* __restrict__ seg, float* out, int E, int Hn, int Cn)
{ long idx = (long)blockIdx.x * 256 + threadIdx.x; if (idx >= (long)E * Hn) return;
  int e = (int)(idx / Hn), h = (int)(idx % Hn);
  float w = logits[idx] / sws[(long)seg[e] * Hn + h];
  const float* src = &feat[(long)gidx[e] * Hn * Cn + h * Cn];
  float* dst = &out[(long)seg[e] * Hn * Cn + h * Cn];
  for (int c = 0; c < Cn; ++c) atomicAdd(&dst[c], src[c] * w); }

// wave-per-row dot + sigmoid (na/pd heads)
__global__ __launch_bounds__(256)
void k_rowdot_sig(const float* __restrict__ in, const float* __restrict__ w,
                  const float* __restrict__ b, float* out, int n, int K, int plusone)
{ int row = blockIdx.x * 8 + (threadIdx.x >> 5); int lane = threadIdx.x & 31;
  if (row >= n) return; float s = 0.f;
  for (int d = lane; d < K; d += 32) s += in[(long)row * K + d] * w[d];
#pragma unroll
  for (int m = 16; m >= 1; m >>= 1) s += __shfl_xor(s, m, 32);
  if (lane == 0) { float sg = 1.f / (1.f + expf(-(s + b[0])));
      out[row] = plusone ? 1.f + sg : sg; } }

__global__ __launch_bounds__(256)
void k_dual_dot(const float* __restrict__ in, const float* __restrict__ w1,
                const float* __restrict__ w2, float* o1, float* o2, int n)
{ int row = blockIdx.x * 8 + (threadIdx.x >> 5); int lane = threadIdx.x & 31;
  if (row >= n) return; float s1 = 0.f, s2 = 0.f;
  for (int d = lane; d < DD; d += 32) { float v = in[(long)row * DD + d];
      s1 += v * w1[d]; s2 += v * w2[d]; }
#pragma unroll
  for (int m = 16; m >= 1; m >>= 1) { s1 += __shfl_xor(s1, m, 32); s2 += __shfl_xor(s2, m, 32); }
  if (lane == 0) { o1[row] = s1; o2[row] = s2; } }

// pooling
__global__ void k_pool_cnt(const int* batch, float* cnt, int n)
{ int i = blockIdx.x * 256 + threadIdx.x; if (i < n) atomicAdd(&cnt[batch[i]], 1.f); }
__global__ void k_pool_sum(const float* h, const int* batch, float* psum, int n)
{ long idx = (long)blockIdx.x * 256 + threadIdx.x; if (idx >= (long)n * DD) return;
  int i = (int)(idx / DD), d = (int)(idx % DD);
  atomicAdd(&psum[(long)batch[i] * DD + d], h[idx]); }
__global__ void k_pool_max(const float* h, const int* batch, unsigned* pmax, int n)
{ long idx = (long)blockIdx.x * 256 + threadIdx.x; if (idx >= (long)n * DD) return;
  int i = (int)(idx / DD), d = (int)(idx % DD);
  atomicMax(&pmax[(long)batch[i] * DD + d], fenc(h[idx])); }
__global__ void k_build_pooled(const float* psum, const unsigned* pmax,
                               const float* cnt, float* pooled)
{ int idx = blockIdx.x * 256 + threadIdx.x; if (idx >= NG * DD) return;
  int g = idx / DD, d = idx % DD;
  float s = psum[idx]; float c = fmaxf(cnt[g], 1.f);
  float m = fdec(pmax[idx]); if (!isfinite(m)) m = 0.f;
  pooled[(long)g * 3 * DD + d]          = s / c;
  pooled[(long)g * 3 * DD + DD + d]     = m;
  pooled[(long)g * 3 * DD + 2 * DD + d] = s; }

// ---------------------------------------------------------------------------
// Host-side orchestration
// ---------------------------------------------------------------------------
static inline dim3 g1(long n, int b = 256) { return dim3((unsigned)((n + b - 1) / b)); }

static void gemm(hipStream_t s, const float* A, const bf16_t* W, const float* bias,
                 float* out, int rows, int K, int Kpad, int NT, int flags)
{
    dim3 grid((rows + 127) / 128), block(256);
    switch (NT) {
    case 4:  k_gemm<4><<<grid, block, 0, s>>>(A, W, bias, out, rows, K, Kpad, flags); break;
    case 6:  k_gemm<6><<<grid, block, 0, s>>>(A, W, bias, out, rows, K, Kpad, flags); break;
    default: k_gemm<12><<<grid, block, 0, s>>>(A, W, bias, out, rows, K, Kpad, flags); break;
    }
}

extern "C" void kernel_launch(void* const* d_in, const int* in_sizes, int n_in,
                              void* d_out, int out_size, void* d_ws, size_t ws_size,
                              hipStream_t stream)
{
    (void)in_sizes; (void)n_in; (void)out_size; (void)ws_size;

    const float* x     = (const float*)d_in[0];
    const int*   ei    = (const int*)  d_in[1];
    const float* eattr = (const float*)d_in[2];
    const int*   batch = (const int*)  d_in[3];
    const float* enc_W = (const float*)d_in[4];
    const float* enc_b = (const float*)d_in[5];
    const float* msW[3] = {(const float*)d_in[6], (const float*)d_in[8], (const float*)d_in[10]};
    const float* msb[3] = {(const float*)d_in[7], (const float*)d_in[9], (const float*)d_in[11]};
    const float* ms_pW = (const float*)d_in[12];
    const float* ms_pb = (const float*)d_in[13];
    const float* gat_W = (const float*)d_in[14];
    const float* gat_b = (const float*)d_in[15];
    const float* gat_as = (const float*)d_in[16];
    const float* gat_ad = (const float*)d_in[17];
    const float* gat_ae = (const float*)d_in[18];
    const float* gat_eW = (const float*)d_in[19];
    const float* na_W1 = (const float*)d_in[20];
    const float* na_b1 = (const float*)d_in[21];
    const float* na_W2 = (const float*)d_in[22];
    const float* na_b2 = (const float*)d_in[23];
    const float* pk_W  = (const float*)d_in[24];
    const float* pk_att = (const float*)d_in[25];
    const float* pk_b  = (const float*)d_in[26];
    const float* pd_W1 = (const float*)d_in[27];
    const float* pd_b1 = (const float*)d_in[28];
    const float* pd_W2 = (const float*)d_in[29];
    const float* pd_b2 = (const float*)d_in[30];
    const float *rbW1[4], *rbb1[4], *rbW2[4], *rbb2[4];
    for (int l = 0; l < 4; ++l) {
        rbW1[l] = (const float*)d_in[31 + 4 * l]; rbb1[l] = (const float*)d_in[32 + 4 * l];
        rbW2[l] = (const float*)d_in[33 + 4 * l]; rbb2[l] = (const float*)d_in[34 + 4 * l];
    }
    const float* mp_W = (const float*)d_in[47];
    const float* mp_b = (const float*)d_in[48];

    const int* row = ei;       // edge sources
    const int* col = ei + NE;  // edge destinations

    // workspace bump allocator (floats, 32B-aligned sub-allocs)
    float* base = (float*)d_ws;
    size_t off = 0;
    auto alloc = [&](size_t nf) {
        off = (off + 7) & ~(size_t)7;
        float* p = base + off; off += nf; return p;
    };
    auto allocb = [&](size_t nbf) {
        off = (off + 7) & ~(size_t)7;
        bf16_t* p = (bf16_t*)(base + off); off += (nbf + 1) / 2; return p;
    };

    float* dinv = alloc(NN);
    float* norm = alloc(NE);
    float* h    = alloc((size_t)NN * DD);
    float* t0   = alloc((size_t)NN * DD);
    float* t1   = alloc((size_t)NN * DD);
    float* t2   = alloc((size_t)NN * DD);
    float* u0   = alloc((size_t)NN * 96);
    float* u1   = alloc((size_t)NN * 96);
    float* asrc = alloc((size_t)NN * NH);
    float* adst = alloc((size_t)NN * NH);
    unsigned* mws = (unsigned*)alloc((size_t)NN * NH);
    float* sws  = alloc((size_t)NN * NH);
    float* elog = alloc((size_t)NE * NH);
    float* aedot = alloc(64);
    float* gate = alloc(NN);
    float* pw1  = alloc(NN);
    float* bsrc = alloc(NN);
    float* bdst = alloc(NN);
    float* cnt  = alloc(NG);
    float* psum = alloc((size_t)NG * DD);
    unsigned* pmax = (unsigned*)alloc((size_t)NG * DD);
    float* pooled = alloc((size_t)NG * 3 * DD);

    // bf16 weight images (zero-padded K -> multiple of 32)
    bf16_t* w_enc = allocb((size_t)32 * DD);
    bf16_t* w_ms[3]; for (int i = 0; i < 3; ++i) w_ms[i] = allocb((size_t)DD * 64);
    bf16_t* w_pW  = allocb((size_t)DD * DD);
    bf16_t* w_gat = allocb((size_t)DD * DD);
    bf16_t* w_na1 = allocb((size_t)DD * 96);
    bf16_t* w_pk  = allocb((size_t)DD * DD);
    bf16_t* w_pd1 = allocb((size_t)DD * 96);
    bf16_t *w_rb1[4], *w_rb2[4];
    for (int l = 0; l < 4; ++l) { w_rb1[l] = allocb((size_t)DD * DD); w_rb2[l] = allocb((size_t)DD * DD); }
    bf16_t* w_mp  = allocb((size_t)576 * DD);

    auto cvtw = [&](const float* W, bf16_t* dst, int K, int Kpad, int DOUT) {
        k_cvtw<<<g1((long)Kpad * DOUT), 256, 0, stream>>>(W, dst, K, Kpad, DOUT);
    };
    cvtw(enc_W, w_enc, 12, 32, DD);
    for (int i = 0; i < 3; ++i) cvtw(msW[i], w_ms[i], DD, DD, 64);
    cvtw(ms_pW, w_pW, DD, DD, DD);
    cvtw(gat_W, w_gat, DD, DD, DD);
    cvtw(na_W1, w_na1, DD, DD, 96);
    cvtw(pk_W,  w_pk,  DD, DD, DD);
    cvtw(pd_W1, w_pd1, DD, DD, 96);
    for (int l = 0; l < 4; ++l) { cvtw(rbW1[l], w_rb1[l], DD, DD, DD); cvtw(rbW2[l], w_rb2[l], DD, DD, DD); }
    cvtw(mp_W, w_mp, 3 * DD, 3 * DD, DD);

    auto prop = [&](const float* in, float* out, int width) {
        k_prop_init<<<g1((long)NN * width), 256, 0, stream>>>(out, in, dinv, NN, width);
        k_prop_edges<<<dim3(NE), dim3(width), 0, stream>>>(out, in, row, col, norm, width);
    };

    // 1) GCN normalization
    k_fill_f32<<<g1(NN), 256, 0, stream>>>(dinv, 0.f, NN);
    k_deg<<<g1(NE), 256, 0, stream>>>(dinv, col, eattr, NE);
    k_dinv<<<g1(NN), 256, 0, stream>>>(dinv, NN);
    k_norm<<<g1(NE), 256, 0, stream>>>(norm, row, col, eattr, dinv, NE);

    // 2) encoder: h = gelu(ln(x @ enc_W + enc_b))
    gemm(stream, x, w_enc, enc_b, h, NN, 12, 32, 12, GF_LN | GF_GELU);

    // 3) multi-scale branches, fused concat-GEMM via accumulate
    static const int scales[3] = {1, 2, 4};
    for (int i = 0; i < 3; ++i) {
        gemm(stream, h, w_ms[i], nullptr, u0, NN, DD, DD, 4, 0);  // N x 64
        prop(u0, u1, 64);
        k_rowops<<<g1((long)NN, 8), 256, 0, stream>>>(u1, u1, msb[i], nullptr, nullptr,
                                                      NN, 64, RF_BIAS);
        float* cur = u1; float* alt = u0;
        for (int r = 1; r < scales[i]; ++r) { prop(cur, alt, 64); float* t = cur; cur = alt; alt = t; }
        int fl = (i == 0) ? 0 : GF_ACC;
        if (i == 2) fl |= GF_LN | GF_GELU;
        gemm(stream, cur, w_pW + (size_t)i * 64 * DD, (i == 0) ? ms_pb : nullptr,
             t0, NN, 64, 64, 12, fl);
    }
    // t0 = post-multiscale hidden

    // 4) GAT layer
    gemm(stream, t0, w_gat, nullptr, t1, NN, DD, DD, 12, 0);     // t1 = xp
    k_head_dots<<<g1((long)NN * NH), 256, 0, stream>>>(t1, gat_as, gat_ad, asrc, adst, NN);
    k_aedot<<<1, 64, 0, stream>>>(gat_eW, gat_ae, aedot);
    k_fill_u32<<<g1((long)NN * NH), 256, 0, stream>>>(mws, ENC_NEG_INF, (long)NN * NH);
    k_fill_f32<<<g1((long)NN * NH), 256, 0, stream>>>(sws, 0.f, (long)NN * NH);
    k_fill_f32<<<g1((long)NN * DD), 256, 0, stream>>>(t2, 0.f, (long)NN * DD);
    k_edge_logits<<<g1((long)NE * NH), 256, 0, stream>>>(asrc, adst, eattr, aedot,
                                                         row, col, col, mws, elog, NE, NH);
    k_edge_exp<<<g1((long)NE * NH), 256, 0, stream>>>(elog, mws, col, sws, NE, NH);
    k_edge_agg<<<g1((long)NE * NH), 256, 0, stream>>>(t1, elog, sws, row, col, t2, NE, NH, CC);
    k_rowops<<<g1((long)NN, 8), 256, 0, stream>>>(t2, t2, gat_b, nullptr, nullptr,
                                                  NN, DD, RF_BIAS);          // t2 = out
    // gating: gate = sigmoid(gelu(out@na_W1+b1)@na_W2+b2); h = ln(out*gate)
    gemm(stream, t2, w_na1, na_b1, u0, NN, DD, DD, 6, GF_GELU);
    k_rowdot_sig<<<g1((long)NN, 8), 256, 0, stream>>>(u0, na_W2, na_b2, gate, NN, 96, 0);
    k_rowops<<<g1((long)NN, 8), 256, 0, stream>>>(t2, h, nullptr, gate, nullptr,
                                                  NN, DD, RF_MUL | RF_LN);

    // 5) peak (pk) attention layer
    gemm(stream, h, w_pk, nullptr, t0, NN, DD, DD, 12, 0);       // t0 = xl
    gemm(stream, t0, w_pd1, pd_b1, u0, NN, DD, DD, 6, GF_GELU);
    k_rowdot_sig<<<g1((long)NN, 8), 256, 0, stream>>>(u0, pd_W2, pd_b2, pw1, NN, 96, 1);
    k_dual_dot<<<g1((long)NN, 8), 256, 0, stream>>>(t0, pk_att, pk_att + DD, bsrc, bdst, NN);
    k_fill_u32<<<g1(NN), 256, 0, stream>>>(mws, ENC_NEG_INF, NN);
    k_fill_f32<<<g1(NN), 256, 0, stream>>>(sws, 0.f, NN);
    k_fill_f32<<<g1((long)NN * DD), 256, 0, stream>>>(t1, 0.f, (long)NN * DD);
    k_edge_logits<<<g1(NE), 256, 0, stream>>>(bsrc, bdst, nullptr, nullptr,
                                              row, col, row, mws, elog, NE, 1);
    k_edge_exp<<<g1(NE), 256, 0, stream>>>(elog, mws, row, sws, NE, 1);
    k_edge_agg<<<g1(NE), 256, 0, stream>>>(t0, elog, sws, row, row, t1, NE, 1, DD);
    k_rowops<<<g1((long)NN, 8), 256, 0, stream>>>(t1, h, pk_b, pw1, nullptr,
                                                  NN, DD, RF_MUL | RF_BIAS);  // h = agg*(1+pw)+pk_b

    // 6) four GCN residual blocks
    for (int l = 0; l < 4; ++l) {
        gemm(stream, h, w_rb1[l], nullptr, t0, NN, DD, DD, 12, 0);
        prop(t0, t1, DD);
        k_rowops<<<g1((long)NN, 8), 256, 0, stream>>>(t1, t1, rbb1[l], nullptr, nullptr,
                                                      NN, DD, RF_BIAS | RF_LN | RF_GELU);
        gemm(stream, t1, w_rb2[l], nullptr, t0, NN, DD, DD, 12, 0);
        prop(t0, t1, DD);
        k_rowops<<<g1((long)NN, 8), 256, 0, stream>>>(t1, h, rbb2[l], nullptr, h,
                                                      NN, DD, RF_BIAS | RF_LN | RF_RES | RF_GELU);
    }

    // 7) graph pooling (mean | max | sum) + readout
    k_fill_f32<<<g1(NG), 256, 0, stream>>>(cnt, 0.f, NG);
    k_fill_f32<<<g1((long)NG * DD), 256, 0, stream>>>(psum, 0.f, (long)NG * DD);
    k_fill_u32<<<g1((long)NG * DD), 256, 0, stream>>>(pmax, ENC_NEG_INF, (long)NG * DD);
    k_pool_cnt<<<g1(NN), 256, 0, stream>>>(batch, cnt, NN);
    k_pool_sum<<<g1((long)NN * DD), 256, 0, stream>>>(h, batch, psum, NN);
    k_pool_max<<<g1((long)NN * DD), 256, 0, stream>>>(h, batch, pmax, NN);
    k_build_pooled<<<g1((long)NG * DD), 256, 0, stream>>>(psum, pmax, cnt, pooled);
    gemm(stream, pooled, w_mp, mp_b, (float*)d_out, NG, 3 * DD, 3 * DD, 12, GF_LN | GF_GELU);
}
